// LSTMcell_57312043598299
// MI455X (gfx1250) — matrix operand order repaired and checked by reading, not compiled, not run on hardware
//
#include <hip/hip_runtime.h>

#define NWG    32
#define TSTEPS 1024
#define BATCH  64
#define HID    512

typedef __attribute__((ext_vector_type(16))) __bf16       v16bf;
typedef __attribute__((ext_vector_type(8)))  float        v8f;
typedef __attribute__((ext_vector_type(8)))  unsigned int v8u;
typedef __attribute__((ext_vector_type(4)))  unsigned int v4u;

struct U8 { v4u lo, hi; };   // 32B, no padding -> bit_cast to v16bf

__device__ __forceinline__ unsigned bf16w(float f) {
    unsigned u = __float_as_uint(f);
    return ((u + 0x7FFFu + ((u >> 16) & 1u)) >> 16) & 0xFFFFu;  // RNE fp32->bf16
}
__device__ __forceinline__ unsigned packbf(float a, float b) {
    return bf16w(a) | (bf16w(b) << 16);
}

// ---------------------------------------------------------------------------
// One-time: x [T,B,512] fp32 -> packed bf16 pairs (bandwidth-bound, parallel).
// ---------------------------------------------------------------------------
__global__ void x_to_bf16(const float* __restrict__ x, uint2* __restrict__ xbf2, int n2) {
    const int stride = gridDim.x * blockDim.x;
    for (int i = blockIdx.x * blockDim.x + threadIdx.x; i < n2; i += stride) {
        const float4 w = ((const float4*)x)[i];
        xbf2[i] = make_uint2(packbf(w.x, w.y), packbf(w.z, w.w));
    }
}

// ---------------------------------------------------------------------------
// Setup: swizzle W = [Wx;Wh] (1024 x 2048 fp32) into per-WG bf16 B-fragments
// (16x16x32 bf16 WMMA B layout: lane n holds col n; lanes 0-15 K=0..7,16..23,
// lanes 16-31 K=8..15,24..31, K-pairs per VGPR). Also init h buffer 0.
// ---------------------------------------------------------------------------
__global__ void lstm_setup(const float* __restrict__ W_ii, const float* __restrict__ W_hi,
                           const float* __restrict__ W_if, const float* __restrict__ W_hf,
                           const float* __restrict__ W_ig, const float* __restrict__ W_hg,
                           const float* __restrict__ W_io, const float* __restrict__ W_ho,
                           const float* __restrict__ h0,
                           unsigned* __restrict__ Wswz, unsigned* __restrict__ hbuf0) {
    const int wg   = blockIdx.x;
    const int tid  = threadIdx.x;
    const int wave = tid >> 5;
    const int lane = tid & 31;
    const int c0   = wg * 16;
    const int n    = lane & 15;
    const int kh   = lane >> 4;

    for (int f = wave; f < 128; f += 8) {          // kt in [0,32), nt in [0,4)
        const int kt = f >> 2, nt = f & 3;
        const float* Wx = (nt == 0) ? W_ii : (nt == 1) ? W_if : (nt == 2) ? W_ig : W_io;
        const float* Wh = (nt == 0) ? W_hi : (nt == 1) ? W_hf : (nt == 2) ? W_hg : W_ho;
        const int col = c0 + n;
        const int kb  = kt * 32 + kh * 8;
        unsigned* dst = Wswz + ((size_t)(wg * 128 + f) * 32 + lane) * 8;
#pragma unroll
        for (int v = 0; v < 8; ++v) {
            const int k = kb + 2 * (v & 3) + 16 * (v >> 2);
            float a, b;
            if (k < 512) { a = Wx[(size_t)k * HID + col];         b = Wx[(size_t)(k + 1) * HID + col]; }
            else         { a = Wh[(size_t)(k - 512) * HID + col]; b = Wh[(size_t)(k - 511) * HID + col]; }
            dst[v] = packbf(a, b);
        }
    }
    for (int i = wg * 256 + tid; i < BATCH * HID / 2; i += NWG * 256)
        hbuf0[i] = packbf(h0[2 * i], h0[2 * i + 1]);
}

// ---------------------------------------------------------------------------
// Persistent recurrent kernel. Weights live in LDS (128KB slice, loaded once),
// so per step the hot loop is ds_load_b128 (B) + global_load_b128 (A) + wmma
// over 4 independent accumulator chains; next-step x is prefetched
// (global_prefetch_b8) so its HBM/L2 latency overlaps the current step's
// WMMA work. Cell update in LDS, publish h bf16, one grid barrier per step.
// LDS layout (dynamic): wlds[32768] u32 | gates[4096] f32 | cbuf[1024] f32 |
//                       bias[64] f32  -> 151808 bytes (< 320KB WGP limit).
// ---------------------------------------------------------------------------
__global__ void lstm_main(const unsigned* __restrict__ xbf, const float* __restrict__ c0in,
                          const float* __restrict__ b_i, const float* __restrict__ b_f,
                          const float* __restrict__ b_g, const float* __restrict__ b_o,
                          const unsigned* __restrict__ Wswz, unsigned* counter,
                          unsigned* hbuf0, unsigned* hbuf1, float* __restrict__ out) {
    extern __shared__ unsigned smem[];
    unsigned* wlds = smem;                           // 32768 u32 = 128KB
    float*    gates = (float*)(smem + 32768);        // 4096 f32  = 16KB
    float*    cbuf  = (float*)(smem + 32768 + 4096); // 1024 f32
    float*    bias  = (float*)(smem + 32768 + 5120); // 64 f32

    const int wg   = blockIdx.x;
    const int tid  = threadIdx.x;
    const int wave = tid >> 5;
    const int lane = tid & 31;
    const int c0   = wg * 16;

    // ---- one-time: pull this WG's swizzled weight slice into LDS ----
    {
        const v4u* src = (const v4u*)(Wswz + (size_t)wg * 32768);
        v4u*       dst = (v4u*)wlds;
        for (int i = tid; i < 32768 / 4; i += 256) dst[i] = src[i];
    }
    if (tid < 64) {
        const int g = tid >> 4, nn = tid & 15;
        const float* bp = (g == 0) ? b_i : (g == 1) ? b_f : (g == 2) ? b_g : b_o;
        bias[tid] = bp[c0 + nn];
    }
    for (int idx = tid; idx < BATCH * 16; idx += 256) {
        const int row = idx >> 4, nn = idx & 15;
        cbuf[idx] = c0in[(size_t)row * HID + c0 + nn];
    }
    __syncthreads();

    const int mt  = wave >> 1;
    const int nt0 = (wave & 1) * 2, nt1 = nt0 + 1;
    const int nl  = lane & 15;
    const int kh  = lane >> 4;
    const int arow = mt * 16 + nl;                          // batch row for A
    const v8u* wl = ((const v8u*)wlds) + lane;              // frag(kt,nt) at +((kt*4+nt)*32)

    for (int t = 0; t < TSTEPS; ++t) {
        unsigned*       hwrite = (t & 1) ? hbuf0 : hbuf1;
        const unsigned* hread  = (t & 1) ? hbuf1 : hbuf0;

        // Prefetch next step's x A-fragment rows: 2 lanes share a 1KB row,
        // each lane covers its 512B half in 4x128B segments.
        if (t + 1 < TSTEPS) {
            const unsigned* xnext = xbf + ((size_t)(t + 1) * BATCH + arow) * 256 + kh * 128;
            __builtin_prefetch(xnext,      0, 3);
            __builtin_prefetch(xnext + 32, 0, 3);
            __builtin_prefetch(xnext + 64, 0, 3);
            __builtin_prefetch(xnext + 96, 0, 3);
        }

        v8f acc0a = {}, acc0b = {}, acc1a = {}, acc1b = {};
        // A-fragment (16-bit A layout): uints kt*16 + kh*4 + {0..3},
        //                                     kt*16 + 8 + kh*4 + {0..3}
        const unsigned* xrow = xbf + ((size_t)t * BATCH + arow) * 256 + kh * 4;
#pragma unroll 4
        for (int kt = 0; kt < 16; ++kt) {                   // x half: K 0..511
            U8 u;
            u.lo = *(const v4u*)(xrow + kt * 16);
            u.hi = *(const v4u*)(xrow + kt * 16 + 8);
            v16bf A  = __builtin_bit_cast(v16bf, u);
            v16bf B0 = __builtin_bit_cast(v16bf, wl[(kt * 4 + nt0) * 32]);
            v16bf B1 = __builtin_bit_cast(v16bf, wl[(kt * 4 + nt1) * 32]);
            if (kt & 1) {
                acc0b = __builtin_amdgcn_wmma_f32_16x16x32_bf16(false, A, false, B0, (short)0, acc0b, false, false);
                acc1b = __builtin_amdgcn_wmma_f32_16x16x32_bf16(false, A, false, B1, (short)0, acc1b, false, false);
            } else {
                acc0a = __builtin_amdgcn_wmma_f32_16x16x32_bf16(false, A, false, B0, (short)0, acc0a, false, false);
                acc1a = __builtin_amdgcn_wmma_f32_16x16x32_bf16(false, A, false, B1, (short)0, acc1a, false, false);
            }
        }
        const unsigned* hrow = hread + (size_t)arow * 256 + kh * 4;
#pragma unroll 4
        for (int kt = 0; kt < 16; ++kt) {                   // h half: K 512..1023
            U8 u;
            u.lo = *(const v4u*)(hrow + kt * 16);
            u.hi = *(const v4u*)(hrow + kt * 16 + 8);
            v16bf A  = __builtin_bit_cast(v16bf, u);
            v16bf B0 = __builtin_bit_cast(v16bf, wl[((16 + kt) * 4 + nt0) * 32]);
            v16bf B1 = __builtin_bit_cast(v16bf, wl[((16 + kt) * 4 + nt1) * 32]);
            if (kt & 1) {
                acc0b = __builtin_amdgcn_wmma_f32_16x16x32_bf16(false, A, false, B0, (short)0, acc0b, false, false);
                acc1b = __builtin_amdgcn_wmma_f32_16x16x32_bf16(false, A, false, B1, (short)0, acc1b, false, false);
            } else {
                acc0a = __builtin_amdgcn_wmma_f32_16x16x32_bf16(false, A, false, B0, (short)0, acc0a, false, false);
                acc1a = __builtin_amdgcn_wmma_f32_16x16x32_bf16(false, A, false, B1, (short)0, acc1a, false, false);
            }
        }
        const v8f acc0 = acc0a + acc0b;
        const v8f acc1 = acc1a + acc1b;

        // D layout: VGPR v -> row v + kh*8, col = lane&15
        {
            const int m0 = mt * 16 + kh * 8;
#pragma unroll
            for (int v = 0; v < 8; ++v) {
                gates[(m0 + v) * 64 + nt0 * 16 + nl] = acc0[v];
                gates[(m0 + v) * 64 + nt1 * 16 + nl] = acc1[v];
            }
        }
        __syncthreads();

        const bool last = (t == TSTEPS - 1);
        for (int idx = tid; idx < BATCH * 16; idx += 256) {
            const int row = idx >> 4, nn = idx & 15;
            float iv = gates[row * 64 +  0 + nn] + bias[nn];
            float fv = gates[row * 64 + 16 + nn] + bias[16 + nn];
            float gv = gates[row * 64 + 32 + nn] + bias[32 + nn];
            float ov = gates[row * 64 + 48 + nn] + bias[48 + nn];
            iv = 1.0f / (1.0f + __expf(-iv));
            fv = 1.0f / (1.0f + __expf(-fv));
            ov = 1.0f / (1.0f + __expf(-ov));
            gv = tanhf(gv);
            const float c = fv * cbuf[idx] + iv * gv;
            cbuf[idx] = c;
            const float h = ov * tanhf(c);
            ((unsigned short*)hwrite)[(size_t)row * HID + c0 + nn] = (unsigned short)bf16w(h);
            if (last) {
                out[(size_t)row * HID + c0 + nn]               = h;
                out[BATCH * HID + (size_t)row * HID + c0 + nn] = c;
            }
        }

        __syncthreads();
        if (tid == 0) {
            __threadfence();
            atomicAdd(counter, 1u);
            const unsigned want = (unsigned)(t + 1) * NWG;
            while (__hip_atomic_load(counter, __ATOMIC_ACQUIRE, __HIP_MEMORY_SCOPE_AGENT) < want) {}
            __threadfence();
        }
        __syncthreads();
    }
}

extern "C" void kernel_launch(void* const* d_in, const int* in_sizes, int n_in,
                              void* d_out, int out_size, void* d_ws, size_t ws_size,
                              hipStream_t stream) {
    const float* x    = (const float*)d_in[0];
    const float* h0   = (const float*)d_in[1];
    const float* c0   = (const float*)d_in[2];
    const float* W_ii = (const float*)d_in[3];
    const float* W_hi = (const float*)d_in[4];
    const float* b_i  = (const float*)d_in[5];
    const float* W_if = (const float*)d_in[6];
    const float* W_hf = (const float*)d_in[7];
    const float* b_f  = (const float*)d_in[8];
    const float* W_ig = (const float*)d_in[9];
    const float* W_hg = (const float*)d_in[10];
    const float* b_g  = (const float*)d_in[11];
    const float* W_io = (const float*)d_in[12];
    const float* W_ho = (const float*)d_in[13];
    const float* b_o  = (const float*)d_in[14];
    float* out = (float*)d_out;

    // workspace layout
    unsigned* counter = (unsigned*)d_ws;                                   // 256B reserved
    unsigned* hbuf0   = (unsigned*)((char*)d_ws + 256);                    // 64KB
    unsigned* hbuf1   = (unsigned*)((char*)d_ws + 256 + 65536);            // 64KB
    unsigned* Wswz    = (unsigned*)((char*)d_ws + 256 + 2 * 65536);        // 4MB
    unsigned* xbf     = (unsigned*)((char*)d_ws + 256 + 2 * 65536 + (4u << 20)); // 64MB

    hipMemsetAsync(d_ws, 0, 256, stream);   // zero grid-barrier counter

    const int n2 = TSTEPS * BATCH * HID / 4;            // float4 granules
    x_to_bf16<<<4096, 256, 0, stream>>>(x, (uint2*)xbf, n2);
    lstm_setup<<<NWG, 256, 0, stream>>>(W_ii, W_hi, W_if, W_hf, W_ig, W_hg, W_io, W_ho,
                                        h0, Wswz, hbuf0);
    const size_t shmem = (32768 + 4096 + 1024 + 64) * sizeof(unsigned);    // 151808 B
    lstm_main<<<NWG, 256, shmem, stream>>>(xbf, c0, b_i, b_f, b_g, b_o,
                                           Wswz, counter, hbuf0, hbuf1, out);
}
